// GCN_35536559407262
// MI455X (gfx1250) — matrix operand order repaired and checked by reading, not compiled, not run on hardware
//
#include <hip/hip_runtime.h>

// ---------------------------------------------------------------------------
// GCN forward for MI455X (gfx1250):
//   - bf16 pre-conversion passes (x, W1, W2, hidden activations)
//   - GEMMs: pure TDM (tensor_load_to_lds) double-buffered bf16 tiles feeding
//     v_wmma_f32_16x16x32_bf16 -- no conversion/transpose on the WMMA path
//   - edge-parallel SpMM scatter with native f32 atomics (L2-resident)
// ---------------------------------------------------------------------------

typedef __attribute__((ext_vector_type(16))) __bf16          v16bf;
typedef __attribute__((ext_vector_type(8)))  float           v8f;
typedef __attribute__((ext_vector_type(8)))  unsigned short  v8us;
typedef __attribute__((ext_vector_type(4)))  unsigned int    v4u;
typedef __attribute__((ext_vector_type(8)))  int             v8i;
typedef __attribute__((ext_vector_type(4)))  int             v4i;
typedef unsigned short ushort_t;

#define NFEAT  512
#define NHID   256
#define NCLASS 64

#if defined(__has_builtin)
#if __has_builtin(__builtin_amdgcn_tensor_load_to_lds) && \
    __has_builtin(__builtin_amdgcn_s_wait_tensorcnt)
#define USE_TDM 1
#endif
#endif
#ifndef USE_TDM
#define USE_TDM 0
#endif

// fp32 -> bf16 (RNE)
__device__ __forceinline__ unsigned short f2bf(float f) {
  unsigned int u = __float_as_uint(f);
  u += 0x7fffu + ((u >> 16) & 1u);
  return (unsigned short)(u >> 16);
}

// ---------------------------------------------------------------------------
// GEMM: C[M,N] = A[M,K] @ Bt[N,K]^T, all-bf16 operands, fp32 accumulate.
// Block tile 128x64, 8 waves, wave tile 32x32, BK=64 (8 WMMAs per stage).
// A: [M][K] bf16 row-major.  Bt: [N][K] bf16 (pre-transposed weights).
// Grid: blockIdx.x = N tile (fast) so sibling blocks reuse the A strip in L2.
// ---------------------------------------------------------------------------
#define BM 128
#define BN 64
#define BK 64
#define LDST 72   // ushorts per LDS row: 64 data + 8 pad = 144B (16B aligned)

#if USE_TDM
// TDM load of a (tileRows x BK) bf16 tile into LDS (D# per ISA 8.3-8.5).
// dim1 = rows remaining -> TDM zero-fills the M tail.  data_size=2B.
// Pad: after every 32 dwords (one 128B row) insert 4 dwords -> 144B LDS rows.
__device__ __forceinline__ void tdm_load_tile(const ushort_t* g, unsigned ldsOff,
                                              long long rowsRemain, int colsRemain,
                                              int strideElems, int tileRows) {
  unsigned long long ga = (unsigned long long)(const void*)g;
  v4u g0;
  g0[0] = 1u;                                            // count=1, user mode
  g0[1] = ldsOff;                                        // lds_addr (bytes)
  g0[2] = (unsigned)(ga & 0xFFFFFFFFull);                // global_addr lo
  g0[3] = (unsigned)((ga >> 32) & 0x1FFFFFFull)          // global_addr hi
          | (2u << 30);                                  // type=2 ("image")
  unsigned td0 = (unsigned)colsRemain;                   // tensor_dim0 (elems)
  unsigned td1 = (unsigned)rowsRemain;                   // tensor_dim1
  v8i g1;
  g1[0] = (int)((1u << 16)        // data_size = 2B
              | (1u << 20)        // pad_enable
              | (4u << 22)        // pad_interval: 32 dwords
              | (3u << 25));      // pad_amount: 4 dwords
  g1[1] = (int)((td0 & 0xFFFFu) << 16);
  g1[2] = (int)(((td0 >> 16) & 0xFFFFu) | ((td1 & 0xFFFFu) << 16));
  g1[3] = (int)(((td1 >> 16) & 0xFFFFu) | ((unsigned)BK << 16)); // tile_dim0
  g1[4] = tileRows;                                      // tile_dim1 (dim2=0)
  g1[5] = strideElems;                                   // dim0 stride lo32
  g1[6] = 0;
  g1[7] = 0;
  v4i gz4 = {0, 0, 0, 0};
  v8i gz8 = {0, 0, 0, 0, 0, 0, 0, 0};
  __builtin_amdgcn_tensor_load_to_lds(g0, g1, gz4, gz4, gz8, 0);
}
#endif

__global__ __launch_bounds__(256)
void gemm_bf16_wmma(const ushort_t* __restrict__ A, const ushort_t* __restrict__ Bt,
                    float* __restrict__ C, int M, int N, int K) {
  __shared__ ushort_t Af[2][BM * LDST];   // [m][k] bf16, double buffered
  __shared__ ushort_t Bf[2][BN * LDST];   // [n][k] bf16, double buffered

  const int tid  = threadIdx.x;
  const int wave = tid >> 5;
  const int lane = tid & 31;
  const int half = lane >> 4;     // 0: lanes 0-15, 1: lanes 16-31
  const int lr   = lane & 15;
  const int wm   = wave >> 1;     // 0..3 (M)
  const int wn   = wave & 1;      // 0..1 (N)
  const int blockN = blockIdx.x * BN;   // N fast -> L2 reuse of A strip
  const int blockM = blockIdx.y * BM;
  const int nstages = K / BK;

  v8f acc[2][2] = {};

#if USE_TDM
  const unsigned ldsA0 = (unsigned)(unsigned long long)(void*)&Af[0][0];
  const unsigned ldsA1 = (unsigned)(unsigned long long)(void*)&Af[1][0];
  const unsigned ldsB0 = (unsigned)(unsigned long long)(void*)&Bf[0][0];
  const unsigned ldsB1 = (unsigned)(unsigned long long)(void*)&Bf[1][0];
  // prologue: DMA stage 0
  if (wave == 0) {
    tdm_load_tile(A + (size_t)blockM * K, ldsA0, (long long)M - blockM, K, K, BM);
    tdm_load_tile(Bt + (size_t)blockN * K, ldsB0, (long long)N - blockN, K, K, BN);
  }
#endif

  for (int s = 0; s < nstages; ++s) {
    const int cur = s & 1;
#if USE_TDM
    if (wave == 0) {
      if (s + 1 < nstages) {
        const int k1 = (s + 1) * BK;
        tdm_load_tile(A + (size_t)blockM * K + k1, (s & 1) ? ldsA0 : ldsA1,
                      (long long)M - blockM, K - k1, K, BM);
        tdm_load_tile(Bt + (size_t)blockN * K + k1, (s & 1) ? ldsB0 : ldsB1,
                      (long long)N - blockN, K - k1, K, BN);
        __builtin_amdgcn_s_wait_tensorcnt(2);  // stage s done, s+1 in flight
      } else {
        __builtin_amdgcn_s_wait_tensorcnt(0);
      }
    }
#else
    {
      const int k0 = s * BK;
      // A stage: 128x64 bf16 = 1024 16B chunks, 4/thread
#pragma unroll
      for (int i = 0; i < 4; ++i) {
        int c  = tid + i * 256;     // 0..1023
        int row = c >> 3;           // 0..127
        int cc  = (c & 7) << 3;     // 0,8,..,56
        v8us v = {0, 0, 0, 0, 0, 0, 0, 0};
        int gr = blockM + row;
        if (gr < M) v = *(const v8us*)(A + (size_t)gr * K + k0 + cc);
        *(v8us*)&Af[cur][row * LDST + cc] = v;
      }
      // B stage: 64x64 bf16 = 512 16B chunks, 2/thread
#pragma unroll
      for (int i = 0; i < 2; ++i) {
        int c  = tid + i * 256;     // 0..511
        int row = c >> 3;           // 0..63
        int cc  = (c & 7) << 3;
        v8us v = *(const v8us*)(Bt + (size_t)(blockN + row) * K + k0 + cc);
        *(v8us*)&Bf[cur][row * LDST + cc] = v;
      }
    }
#endif
    __syncthreads();

    // 2 K-subtiles of 32 -> 8 WMMAs per stage; pure ds_load + wmma
#pragma unroll
    for (int kk = 0; kk < BK; kk += 32) {
      // A frag (16-bit A 16x32): lane<16 -> K kk+[0..7] & kk+[16..23]
      union { v16bf v; v8us h[2]; } afr[2], bfr[2];
#pragma unroll
      for (int mi = 0; mi < 2; ++mi) {
        int row = wm * 32 + mi * 16 + lr;
        const ushort_t* ap = &Af[cur][row * LDST + kk + half * 8];
        afr[mi].h[0] = *(const v8us*)(ap);
        afr[mi].h[1] = *(const v8us*)(ap + 16);
      }
      // B frag (dense 32x16): lane<16 col=lr holds K kk+[0..7] & kk+[16..23]
#pragma unroll
      for (int ni = 0; ni < 2; ++ni) {
        int col = wn * 32 + ni * 16 + lr;
        const ushort_t* bp = &Bf[cur][col * LDST + kk + half * 8];
        bfr[ni].h[0] = *(const v8us*)(bp);
        bfr[ni].h[1] = *(const v8us*)(bp + 16);
      }
#pragma unroll
      for (int mi = 0; mi < 2; ++mi)
#pragma unroll
        for (int ni = 0; ni < 2; ++ni)
          acc[mi][ni] = __builtin_amdgcn_wmma_f32_16x16x32_bf16(
              false, afr[mi].v, false, bfr[ni].v,
              (short)0, acc[mi][ni], false, false);
    }
    __syncthreads();   // guards next DMA/copy from overwriting live buffers
  }

  // store C: f32 16x16 C/D layout: VGPR r -> row = r + half*8, col = lr
#pragma unroll
  for (int mi = 0; mi < 2; ++mi) {
#pragma unroll
    for (int ni = 0; ni < 2; ++ni) {
      int gcol = blockN + wn * 32 + ni * 16 + lr;
#pragma unroll
      for (int r = 0; r < 8; ++r) {
        int grow = blockM + wm * 32 + mi * 16 + half * 8 + r;
        if (grow < M) C[(size_t)grow * N + gcol] = acc[mi][ni][r];
      }
    }
  }
}

// ---------------------------------------------------------------------------
// Conversion pre-passes
// ---------------------------------------------------------------------------
// fp32[n] -> bf16[n], 8 elements (one b128 store) per thread
__global__ __launch_bounds__(256)
void cvt_bf16(const float* __restrict__ in, ushort_t* __restrict__ out, size_t n8) {
  size_t i = (size_t)blockIdx.x * blockDim.x + threadIdx.x;
  if (i >= n8) return;
  const float* p = in + i * 8;
  float4 a = *(const float4*)(p);
  float4 b = *(const float4*)(p + 4);
  v8us v;
  v[0] = f2bf(a.x); v[1] = f2bf(a.y); v[2] = f2bf(a.z); v[3] = f2bf(a.w);
  v[4] = f2bf(b.x); v[5] = f2bf(b.y); v[6] = f2bf(b.z); v[7] = f2bf(b.w);
  *(v8us*)(out + i * 8) = v;
}

// W[K][N] fp32 -> Wt[N][K] bf16 (tiny: weights only)
__global__ __launch_bounds__(256)
void transpose_cvt(const float* __restrict__ W, ushort_t* __restrict__ Wt,
                   int K, int nshift) {
  const int N = 1 << nshift;
  size_t i = (size_t)blockIdx.x * blockDim.x + threadIdx.x;
  if (i >= ((size_t)K << nshift)) return;
  int k = (int)(i >> nshift);
  int n = (int)(i & (size_t)(N - 1));
  Wt[(size_t)n * K + k] = f2bf(W[i]);
}

// h2 = bf16(relu(agg + b))  -- feeds GEMM2 directly in bf16
__global__ __launch_bounds__(256)
void bias_relu_bf16(const float* __restrict__ agg, const float* __restrict__ b,
                    ushort_t* __restrict__ h2, size_t n, int dmask) {
  size_t i = (size_t)blockIdx.x * blockDim.x + threadIdx.x;
  if (i < n) h2[i] = f2bf(fmaxf(agg[i] + b[i & (size_t)dmask], 0.f));
}

// ---------------------------------------------------------------------------
// Edge-parallel scatter: dst[row_e, :] += w_e * src[col_e, :]
// src fits in 192MB L2 -> gathers and atomics are L2 traffic, not HBM.
// ---------------------------------------------------------------------------
__global__ __launch_bounds__(256)
void spmm_scatter(const int* __restrict__ erow, const int* __restrict__ ecol,
                  const float* __restrict__ ew, const float* __restrict__ src,
                  float* __restrict__ dst, long long E, int d4shift, int D) {
  const size_t total  = (size_t)E << d4shift;
  const size_t stride = (size_t)gridDim.x * blockDim.x;
  for (size_t idx = (size_t)blockIdx.x * blockDim.x + threadIdx.x;
       idx < total; idx += stride) {
    size_t e = idx >> d4shift;
    int    c = (int)(idx & (((size_t)1 << d4shift) - 1)) << 2;
    int    r  = erow[e];
    int    cl = ecol[e];
    float  wt = ew[e];
    float4 v = *(const float4*)(src + (size_t)cl * D + c);
    float* p = dst + (size_t)r * D + c;
    __hip_atomic_fetch_add(p + 0, wt * v.x, __ATOMIC_RELAXED, __HIP_MEMORY_SCOPE_AGENT);
    __hip_atomic_fetch_add(p + 1, wt * v.y, __ATOMIC_RELAXED, __HIP_MEMORY_SCOPE_AGENT);
    __hip_atomic_fetch_add(p + 2, wt * v.z, __ATOMIC_RELAXED, __HIP_MEMORY_SCOPE_AGENT);
    __hip_atomic_fetch_add(p + 3, wt * v.w, __ATOMIC_RELAXED, __HIP_MEMORY_SCOPE_AGENT);
  }
}

__global__ __launch_bounds__(256)
void zero_f32(float* __restrict__ p, size_t n) {
  size_t i = (size_t)blockIdx.x * blockDim.x + threadIdx.x;
  if (i < n) p[i] = 0.f;
}

__global__ __launch_bounds__(256)
void init_bias(float* __restrict__ out, const float* __restrict__ b,
               size_t n, int dmask) {
  size_t i = (size_t)blockIdx.x * blockDim.x + threadIdx.x;
  if (i < n) out[i] = b[i & (size_t)dmask];
}

// ---------------------------------------------------------------------------
extern "C" void kernel_launch(void* const* d_in, const int* in_sizes, int n_in,
                              void* d_out, int out_size, void* d_ws, size_t ws_size,
                              hipStream_t stream) {
  const float* x    = (const float*)d_in[0];
  const int*   erow = (const int*)  d_in[1];
  const int*   ecol = (const int*)  d_in[2];
  const float* ew   = (const float*)d_in[3];
  const float* W1   = (const float*)d_in[4];
  const float* b1   = (const float*)d_in[5];
  const float* W2   = (const float*)d_in[6];
  const float* b2   = (const float*)d_in[7];
  float* out = (float*)d_out;

  const int M = in_sizes[0] / NFEAT;       // 100000 nodes
  const long long E = in_sizes[1];         // 3.2M edges

  // workspace layout (agg1 aliases xbf: xbf is dead once GEMM1 completes)
  char* ws = (char*)d_ws;
  const size_t szH1  = (size_t)M * NHID * 4;     // h1 fp32
  const size_t szXbf = (size_t)M * NFEAT * 2;    // xbf bf16 (== agg1 size)
  const size_t szO   = (size_t)M * NCLASS * 4;   // o fp32
  const size_t szH2  = (size_t)M * NHID * 2;     // h2 bf16
  const size_t szW1t = (size_t)NHID * NFEAT * 2;
  float*    h1   = (float*)   (ws);
  ushort_t* xbf  = (ushort_t*)(ws + szH1);
  float*    agg1 = (float*)   (ws + szH1);            // alias of xbf
  float*    o    = (float*)   (ws + szH1 + szXbf);
  ushort_t* h2   = (ushort_t*)(ws + szH1 + szXbf + szO);
  ushort_t* W1t  = (ushort_t*)(ws + szH1 + szXbf + szO + szH2);
  ushort_t* W2t  = (ushort_t*)(ws + szH1 + szXbf + szO + szH2 + szW1t);

  const size_t nh = (size_t)M * NHID;
  const size_t no = (size_t)M * NCLASS;

  // 0) conversion pre-passes: x -> bf16, W1/W2 -> transposed bf16
  {
    size_t n8 = (size_t)M * NFEAT / 8;
    cvt_bf16<<<(unsigned)((n8 + 255) / 256), 256, 0, stream>>>(x, xbf, n8);
    transpose_cvt<<<(NFEAT * NHID + 255) / 256, 256, 0, stream>>>(W1, W1t, NFEAT, 8);
    transpose_cvt<<<(NHID * NCLASS + 255) / 256, 256, 0, stream>>>(W2, W2t, NHID, 6);
  }

  // 1) h1 = xbf @ W1t^T   (grid: N tiles fast for L2 reuse of A strip)
  dim3 g1(NHID / BN, (M + BM - 1) / BM);
  gemm_bf16_wmma<<<g1, 256, 0, stream>>>(xbf, W1t, h1, M, NHID, NFEAT);

  // 2) agg1 = 0 (reuses xbf storage; xbf dead after GEMM1)
  zero_f32<<<(unsigned)((nh + 255) / 256), 256, 0, stream>>>(agg1, nh);

  // 3) agg1[r] += w_e * h1[c]  (D=256 -> 64 float4 chunks, shift=6)
  {
    size_t total = (size_t)E << 6;
    spmm_scatter<<<(unsigned)((total + 255) / 256), 256, 0, stream>>>(
        erow, ecol, ew, h1, agg1, E, 6, NHID);
  }

  // 4) h2 = bf16(relu(agg1 + b1))
  bias_relu_bf16<<<(unsigned)((nh + 255) / 256), 256, 0, stream>>>(
      agg1, b1, h2, nh, NHID - 1);

  // 5) o = h2 @ W2t^T
  dim3 g2(NCLASS / BN, (M + BM - 1) / BM);
  gemm_bf16_wmma<<<g2, 256, 0, stream>>>(h2, W2t, o, M, NCLASS, NHID);

  // 6) out = b2 (broadcast)
  init_bias<<<(unsigned)((no + 255) / 256), 256, 0, stream>>>(out, b2, no, NCLASS - 1);

  // 7) out[r] += w_e * o[c]  (D=64 -> 16 float4 chunks, shift=4)
  {
    size_t total = (size_t)E << 4;
    spmm_scatter<<<(unsigned)((total + 255) / 256), 256, 0, stream>>>(
        erow, ecol, ew, o, out, E, 4, NCLASS);
  }
}
